// AxialAttention_80711025426885
// MI455X (gfx1250) — matrix operand order repaired and checked by reading, not compiled, not run on hardware
//
#include <hip/hip_runtime.h>

typedef __attribute__((ext_vector_type(16))) __bf16 v16bf;
typedef __attribute__((ext_vector_type(8)))  float  v8f;

struct U128x2 { uint4 a, b; };

static __device__ __forceinline__ v16bf mkv(uint4 a, uint4 b) {
    U128x2 t; t.a = a; t.b = b;
    return __builtin_bit_cast(v16bf, t);
}

// float -> bf16 round-to-nearest-even (inputs are finite normals)
static __device__ __forceinline__ unsigned short f2bf(float f) {
    unsigned int u = __builtin_bit_cast(unsigned int, f);
    u += 0x7FFFu + ((u >> 16) & 1u);
    return (unsigned short)(u >> 16);
}

#define LDA_X  520   // 512 + 8 pad -> row stride 1040B, conflict-free b128 column access
#define LDA_QK 72    // 64 + 8 pad
#define LDA_P  136   // 128 + 8 pad
#define Q_OFF  0
#define K_OFF  9216      // 128*72
#define P_OFF  0         // P overlays q+k (dead after S is computed)
#define VT_OFF 18432     // 2*128*72

__global__ __launch_bounds__(256, 1) void axial_attn_fused(
    const float* __restrict__ x,
    const unsigned short* __restrict__ wq,
    const unsigned short* __restrict__ wk,
    const unsigned short* __restrict__ wv,
    const unsigned short* __restrict__ wo,
    const float* __restrict__ bo,
    float* __restrict__ out)
{
    __shared__ alignas(16) unsigned short lds_x[128 * LDA_X];    // 133120 B
    __shared__ alignas(16) unsigned short lds_ctx[128 * LDA_X];  // 133120 B
    __shared__ alignas(16) unsigned short lds_s[27136];          //  54272 B

    const int tid  = threadIdx.x;
    const int wave = tid >> 5;
    const int lane = tid & 31;
    const int half = lane >> 4;
    const int l16  = lane & 15;

    const int nb = blockIdx.x >> 7;    // batch index
    const int w  = blockIdx.x & 127;   // width coordinate
    const float* xb = x + (nb * 16384 + w) * 512;  // + h*65536 + c

    // Stage x[n,:,w,:] -> LDS as bf16 (coalesced f32 reads)
    for (int idx = tid; idx < 128 * 512; idx += 256) {
        int h = idx >> 9, c = idx & 511;
        lds_x[h * LDA_X + c] = f2bf(xb[h * 65536 + c]);
    }
    __syncthreads();

    for (int g = 0; g < 8; ++g) {
        const int gbase = g * 64;

        // ---- Q,K head projections: 2 mats x 8 mtiles x 4 ntiles = 64 jobs, 8/wave ----
        for (int j = 0; j < 8; ++j) {
            int job = wave * 8 + j;
            int mat = job >> 5;
            int mtile = (job >> 2) & 7;
            int ntile = job & 3;
            const unsigned short* Wg = mat ? wk : wq;
            const unsigned short* arow = lds_x + (mtile * 16 + l16) * LDA_X + half * 8;
            const unsigned short* brow = Wg + (gbase + ntile * 16 + l16) * 512 + half * 8;
            v8f acc = {};
            #pragma unroll 4
            for (int ks = 0; ks < 16; ++ks) {
                int kb = ks * 32;
                v16bf a = mkv(*(const uint4*)(arow + kb), *(const uint4*)(arow + kb + 16));
                v16bf b = mkv(*(const uint4*)(brow + kb), *(const uint4*)(brow + kb + 16));
                acc = __builtin_amdgcn_wmma_f32_16x16x32_bf16(false, a, false, b, (short)0, acc, false, false);
            }
            unsigned short* dst = lds_s + (mat ? K_OFF : Q_OFF);
            float scale = mat ? 0.125f : 1.0f;   // fold 1/sqrt(64) into k
            #pragma unroll
            for (int r = 0; r < 8; ++r) {
                int m = mtile * 16 + half * 8 + r;
                dst[m * LDA_QK + ntile * 16 + l16] = f2bf(acc[r] * scale);
            }
        }
        __syncthreads();

        // ---- S = q @ k^T : each wave owns one 16-row stripe, all 128 cols in regs ----
        v8f s[8];
        #pragma unroll
        for (int t = 0; t < 8; ++t) s[t] = (v8f){};
        {
            const unsigned short* arow = lds_s + Q_OFF + (wave * 16 + l16) * LDA_QK + half * 8;
            #pragma unroll
            for (int ks = 0; ks < 2; ++ks) {
                int kb = ks * 32;
                v16bf a = mkv(*(const uint4*)(arow + kb), *(const uint4*)(arow + kb + 16));
                #pragma unroll
                for (int t = 0; t < 8; ++t) {
                    const unsigned short* bp = lds_s + K_OFF + (t * 16 + l16) * LDA_QK + kb + half * 8;
                    v16bf b = mkv(*(const uint4*)bp, *(const uint4*)(bp + 16));
                    s[t] = __builtin_amdgcn_wmma_f32_16x16x32_bf16(false, a, false, b, (short)0, s[t], false, false);
                }
            }
        }
        // Row softmax: row m lives in 16 lanes of one half-wave across 8 accumulators
        #pragma unroll
        for (int r = 0; r < 8; ++r) {
            float mx = -3.4e38f;
            #pragma unroll
            for (int t = 0; t < 8; ++t) mx = fmaxf(mx, s[t][r]);
            #pragma unroll
            for (int off = 8; off >= 1; off >>= 1) mx = fmaxf(mx, __shfl_xor(mx, off, 32));
            float sum = 0.f;
            #pragma unroll
            for (int t = 0; t < 8; ++t) { float e = __expf(s[t][r] - mx); s[t][r] = e; sum += e; }
            #pragma unroll
            for (int off = 8; off >= 1; off >>= 1) sum += __shfl_xor(sum, off, 32);
            float inv = 1.f / sum;
            #pragma unroll
            for (int t = 0; t < 8; ++t) s[t][r] *= inv;
        }
        __syncthreads();   // all waves done reading q,k -> safe to overwrite with P
        #pragma unroll
        for (int r = 0; r < 8; ++r) {
            int m = wave * 16 + half * 8 + r;
            #pragma unroll
            for (int t = 0; t < 8; ++t)
                lds_s[P_OFF + m * LDA_P + t * 16 + l16] = f2bf(s[t][r]);
        }

        // ---- V projection (store transposed so PV B-operand is contiguous) ----
        for (int j = 0; j < 4; ++j) {
            int job = wave * 4 + j;
            int mtile = job >> 2, ntile = job & 3;
            const unsigned short* arow = lds_x + (mtile * 16 + l16) * LDA_X + half * 8;
            const unsigned short* brow = wv + (gbase + ntile * 16 + l16) * 512 + half * 8;
            v8f acc = {};
            #pragma unroll 4
            for (int ks = 0; ks < 16; ++ks) {
                int kb = ks * 32;
                v16bf a = mkv(*(const uint4*)(arow + kb), *(const uint4*)(arow + kb + 16));
                v16bf b = mkv(*(const uint4*)(brow + kb), *(const uint4*)(brow + kb + 16));
                acc = __builtin_amdgcn_wmma_f32_16x16x32_bf16(false, a, false, b, (short)0, acc, false, false);
            }
            #pragma unroll
            for (int r = 0; r < 8; ++r) {
                int m = mtile * 16 + half * 8 + r;
                lds_s[VT_OFF + (ntile * 16 + l16) * LDA_P + m] = f2bf(acc[r]);
            }
        }
        __syncthreads();

        // ---- ctx_g = P @ v : head g writes disjoint columns of ctx ----
        for (int j = 0; j < 4; ++j) {
            int job = wave * 4 + j;
            int mtile = job >> 2, ntile = job & 3;
            const unsigned short* arow = lds_s + P_OFF + (mtile * 16 + l16) * LDA_P + half * 8;
            const unsigned short* brow = lds_s + VT_OFF + (ntile * 16 + l16) * LDA_P + half * 8;
            v8f acc = {};
            #pragma unroll
            for (int ks = 0; ks < 4; ++ks) {
                int kb = ks * 32;
                v16bf a = mkv(*(const uint4*)(arow + kb), *(const uint4*)(arow + kb + 16));
                v16bf b = mkv(*(const uint4*)(brow + kb), *(const uint4*)(brow + kb + 16));
                acc = __builtin_amdgcn_wmma_f32_16x16x32_bf16(false, a, false, b, (short)0, acc, false, false);
            }
            #pragma unroll
            for (int r = 0; r < 8; ++r) {
                int m = mtile * 16 + half * 8 + r;
                lds_ctx[m * LDA_X + gbase + ntile * 16 + l16] = f2bf(acc[r]);
            }
        }
        __syncthreads();
    }

    // ---- out = ctx @ Wo^T + bo, written to [N,H,W,C] with the output transpose ----
    {
        const int mtile = wave;
        const unsigned short* arow = lds_ctx + (mtile * 16 + l16) * LDA_X + half * 8;
        for (int ntile = 0; ntile < 32; ++ntile) {
            const unsigned short* brow = wo + (ntile * 16 + l16) * 512 + half * 8;
            v8f acc = {};
            #pragma unroll 4
            for (int ks = 0; ks < 16; ++ks) {
                int kb = ks * 32;
                v16bf a = mkv(*(const uint4*)(arow + kb), *(const uint4*)(arow + kb + 16));
                v16bf b = mkv(*(const uint4*)(brow + kb), *(const uint4*)(brow + kb + 16));
                acc = __builtin_amdgcn_wmma_f32_16x16x32_bf16(false, a, false, b, (short)0, acc, false, false);
            }
            float bias = bo[ntile * 16 + l16];
            #pragma unroll
            for (int r = 0; r < 8; ++r) {
                int h = mtile * 16 + half * 8 + r;
                out[((nb * 128 + h) * 128 + w) * 512 + ntile * 16 + l16] = acc[r] + bias;
            }
        }
    }
}

// Pre-convert the four 512x512 f32 weight matrices to bf16 in workspace (2 MB, L2-resident)
__global__ __launch_bounds__(256) void convert_w_kernel(
    const float* __restrict__ a, const float* __restrict__ b,
    const float* __restrict__ c, const float* __restrict__ d,
    unsigned short* __restrict__ dst)
{
    int idx = blockIdx.x * 256 + threadIdx.x;   // 4 * 262144 total
    int m = idx >> 18;
    int o = idx & 262143;
    const float* src = (m == 0) ? a : (m == 1) ? b : (m == 2) ? c : d;
    dst[idx] = f2bf(src[o]);
}

extern "C" void kernel_launch(void* const* d_in, const int* in_sizes, int n_in,
                              void* d_out, int out_size, void* d_ws, size_t ws_size,
                              hipStream_t stream) {
    (void)in_sizes; (void)n_in; (void)out_size; (void)ws_size;
    const float* x  = (const float*)d_in[0];
    const float* Wq = (const float*)d_in[1];
    const float* Wk = (const float*)d_in[2];
    const float* Wv = (const float*)d_in[3];
    const float* Wo = (const float*)d_in[4];
    const float* bo = (const float*)d_in[5];
    unsigned short* wbf = (unsigned short*)d_ws;

    convert_w_kernel<<<4096, 256, 0, stream>>>(Wq, Wk, Wv, Wo, wbf);
    axial_attn_fused<<<1024, 256, 0, stream>>>(
        x, wbf, wbf + 262144, wbf + 524288, wbf + 786432, bo, (float*)d_out);
}
